// Attention_46634754900025
// MI455X (gfx1250) — matrix operand order repaired and checked by reading, not compiled, not run on hardware
//
#include <hip/hip_runtime.h>
#include <hip/hip_bf16.h>

// ---------------------------------------------------------------------------
// Windowed attention for MI455X (gfx1250), wave32 + WMMA f16->f32.
// Roofline: ~86 GFLOP total vs ~500MB HBM traffic -> balanced at f16 WMMA
// rates; f32 inputs converted once to f16, all GEMMs on
// v_wmma_f32_16x16x32_f16, f16 intermediates, LDS-staged tiles.
//
// Workspace (f16): Q 32MB, K 64MB, V 64MB, AO 32MB = 192MB total.
// ---------------------------------------------------------------------------

typedef __attribute__((ext_vector_type(16))) _Float16 v16h;
typedef __attribute__((ext_vector_type(8)))  _Float16 v8h;
typedef __attribute__((ext_vector_type(8)))  float    v8f;
typedef int v4i_vs __attribute__((vector_size(16)));   // matches builtin param

#define CIN   256
#define HH    128
#define WWID  128
#define WIN   16
#define NTOK  256   /* tokens per window */
#define NKV   512   /* kv tokens per window (2 sources) */
#define DH    64
#define HEADS 8
#define NWIN  1024  /* 2 * 8 * 8 * 8 */

// ---- async global->LDS copy (gfx1250 GLOBAL_LOAD_ASYNC_TO_LDS_B128) -------
#if defined(__gfx1250__) && \
    __has_builtin(__builtin_amdgcn_global_load_async_to_lds_b128) && \
    __has_builtin(__builtin_amdgcn_s_wait_asynccnt)
#define USE_ASYNC_LDS 1
#else
#define USE_ASYNC_LDS 0
#endif

__device__ __forceinline__ void copy16_to_lds(const _Float16* g, _Float16* l) {
#if USE_ASYNC_LDS
  __builtin_amdgcn_global_load_async_to_lds_b128(
      (__attribute__((address_space(1))) v4i_vs*)(_Float16*)g,
      (__attribute__((address_space(3))) v4i_vs*)l, 0, 0);
#else
  *(uint4*)l = *(const uint4*)g;
#endif
}

__device__ __forceinline__ void async_copy_wait() {
#if USE_ASYNC_LDS
  __builtin_amdgcn_s_wait_asynccnt(0);
#endif
}

__device__ __forceinline__ v8f wmma16(v16h a, v16h b, v8f c) {
  return __builtin_amdgcn_wmma_f32_16x16x32_f16(false, a, false, b,
                                                (short)0, c, false, false);
}

// A-matrix 16x32 f16 fragment (ISA 7.12.2): lane m=lane&15, half=lane>>4.
// VGPR v (0..7), elem e: k = (v>>2)*16 + half*8 + (v&3)*2 + e.
// Source row-major [M x ld] f16: element(row m, col k) = base[m*ld + k].
__device__ __forceinline__ v16h load_a_frag_f16(const _Float16* base, int ld,
                                                int m, int half, int k0) {
  v16h a;
#pragma unroll
  for (int v = 0; v < 8; ++v) {
    int k = k0 + ((v >> 2) << 4) + (half << 3) + ((v & 3) << 1);
    const _Float16* p = base + (size_t)m * ld + k;
    a[2 * v]     = p[0];
    a[2 * v + 1] = p[1];
  }
  return a;
}

// Same A fragment, converting from f32 weights streamed through L2.
__device__ __forceinline__ v16h load_a_frag_f32(const float* base, int ld,
                                                int m, int half, int k0) {
  v16h a;
#pragma unroll
  for (int v = 0; v < 8; ++v) {
    int k = k0 + ((v >> 2) << 4) + (half << 3) + ((v & 3) << 1);
    const float* p = base + (size_t)m * ld + k;
    a[2 * v]     = (_Float16)p[0];
    a[2 * v + 1] = (_Float16)p[1];
  }
  return a;
}

// B-matrix 32x16 f16 fragment: lane n=lane&15, half=lane>>4.
// VGPR v, elem e: k = (v>>2)*16 + half*8 + (v&3)*2 + e.
// "k-contiguous" storage: element(k, n) = base[n*ld + k].
__device__ __forceinline__ v16h load_b_frag_f16(const _Float16* base, int ld,
                                                int n, int half, int k0) {
  v16h b;
#pragma unroll
  for (int v = 0; v < 8; ++v) {
    int k = k0 + ((v >> 2) << 4) + (half << 3) + ((v & 3) << 1);
    const _Float16* p = base + (size_t)n * ld + k;
    b[2 * v]     = p[0];
    b[2 * v + 1] = p[1];
  }
  return b;
}

// B fragment from "n-contiguous" storage: element(k, n) = base[k*ld + n].
__device__ __forceinline__ v16h load_b_frag_f16_rm(const _Float16* base, int ld,
                                                   int n, int half, int k0) {
  v16h b;
#pragma unroll
  for (int v = 0; v < 8; ++v) {
    int k = k0 + ((v >> 2) << 4) + (half << 3) + ((v & 3) << 1);
    b[2 * v]     = base[(size_t)k * ld + n];
    b[2 * v + 1] = base[(size_t)(k + 1) * ld + n];
  }
  return b;
}

// ---------------------------------------------------------------------------
// Kernel 1: Q/K/V projection. Block = (src in 0..3, window in 0..63).
// src 0,1 = x batches (r=0, also produces Q); src 2,3 = skip batches (r=1).
// LDS: input window token-major f16 [256 tok][258 pitch] = 132KB dynamic.
// ---------------------------------------------------------------------------
#define K1_PITCH 258

__global__ void qkv_proj_kernel(const float* __restrict__ x,
                                const float* __restrict__ skip,
                                const float* __restrict__ Wq,
                                const float* __restrict__ Wkv,
                                _Float16* __restrict__ Qws,
                                _Float16* __restrict__ Kws,
                                _Float16* __restrict__ Vws) {
  extern __shared__ _Float16 lds[];
  const int bx   = blockIdx.x;
  const int src  = bx >> 6;
  const int win  = bx & 63;
  const int Xb   = win >> 3, Yb = win & 7;
  const int b    = src & 1;
  const int rsrc = src >> 1;
  const float* sp = (rsrc == 0 ? x : skip) + (size_t)b * CIN * HH * WWID;

  const int tid = threadIdx.x;
  // Stage window: 256 channels x 256 tokens, f32 -> f16, token-major LDS.
#pragma unroll 4
  for (int it = 0; it < 256; ++it) {
    int idx = it * 256 + tid;
    int c   = idx >> 8;
    int tok = idx & 255;
    int hh  = Xb * WIN + (tok >> 4);
    int ww  = Yb * WIN + (tok & 15);
    float v = sp[(size_t)c * (HH * WWID) + (size_t)hh * WWID + ww];
    lds[(size_t)tok * K1_PITCH + c] = (_Float16)v;
  }
  __syncthreads();

  const int lane = tid & 31;
  const int waveId = tid >> 5;
  const int m = lane & 15;
  const int half = lane >> 4;

  const int mtiles = (rsrc == 0) ? 96 : 64;  // Q(32)+K(32)+V(32) or K+V only
  for (int mt = waveId; mt < mtiles; mt += 8) {
    int kind, och;  // kind: 0=Q 1=K 2=V ; och = output channel base
    if (rsrc == 0) {
      if (mt < 32)      { kind = 0; och = mt * 16; }
      else if (mt < 64) { kind = 1; och = (mt - 32) * 16; }
      else              { kind = 2; och = (mt - 64) * 16; }
    } else {
      if (mt < 32)      { kind = 1; och = mt * 16; }
      else              { kind = 2; och = (mt - 32) * 16; }
    }
    const float* wbase = (kind == 0) ? (Wq  + (size_t)och * CIN)
                       : (kind == 1) ? (Wkv + (size_t)och * CIN)
                                     : (Wkv + (size_t)(512 + och) * CIN);
    __builtin_prefetch(wbase + (size_t)m * CIN, 0, 1);

    // Preload all 8 K-chunk A fragments of this 16-row weight tile.
    v16h afr[8];
#pragma unroll
    for (int kc = 0; kc < 8; ++kc)
      afr[kc] = load_a_frag_f32(wbase, CIN, m, half, kc * 32);

    const int head = och >> 6;
    const int d0   = och & 63;
    const size_t w = (size_t)(((b * 8 + head) * 8 + Xb) * 8 + Yb);

    // Uniform destination select, hoisted out of the tile loops.
    _Float16* dstbase;
    int jbase;
    if (kind == 0) { dstbase = Qws + w * (size_t)(NTOK * DH); jbase = 0; }
    else {
      dstbase = ((kind == 1) ? Kws : Vws) + w * (size_t)(NKV * DH);
      jbase = rsrc * NTOK;  // token-concat order: x first, skip second
    }

    for (int nt = 0; nt < 16; ++nt) {
      v8f acc = {};
#pragma unroll
      for (int kc = 0; kc < 8; ++kc) {
        v16h bfr = load_b_frag_f16(lds, K1_PITCH, nt * 16 + m, half, kc * 32);
        acc = wmma16(afr[kc], bfr, acc);
      }
      // Lane's 8 C elements are d = d0 + half*8 + r: contiguous -> one b128.
      const int tok = nt * 16 + m;
      v8h pk;
#pragma unroll
      for (int r = 0; r < 8; ++r) pk[r] = (_Float16)acc[r];
      *(v8h*)(dstbase + (size_t)(jbase + tok) * DH + d0 + half * 8) = pk;
    }
  }
}

// ---------------------------------------------------------------------------
// Kernel 2: per-window flash attention. Block = window w (1024 blocks).
// LDS: K [512][72] and V [512][72] (async b128 staged, 16B-aligned rows),
// per-wave P scratch [16][36]. Total ~157KB dynamic LDS.
// ---------------------------------------------------------------------------
#define K2_DP 72
#define K2_PP 36

__global__ void attn_kernel(const _Float16* __restrict__ Qws,
                            const _Float16* __restrict__ Kws,
                            const _Float16* __restrict__ Vws,
                            _Float16* __restrict__ AOws) {
  extern __shared__ _Float16 lds[];
  _Float16* Klds = lds;                      // [j=512][pitch 72]
  _Float16* Vlds = Klds + NKV * K2_DP;       // [j=512][pitch 72]
  _Float16* Plds = Vlds + NKV * K2_DP;       // 8 waves x [16][36]

  const int w   = blockIdx.x;
  const int tid = threadIdx.x;
  const _Float16* Kg = Kws + (size_t)w * NKV * DH;
  const _Float16* Vg = Vws + (size_t)w * NKV * DH;

  // Stage K and V [j][d] with 16B async copies (8 f16 per lane per op).
#pragma unroll
  for (int it = 0; it < 16; ++it) {
    int idx = (it * 256 + tid) * 8;
    int j = idx >> 6, d = idx & 63;
    copy16_to_lds(Kg + idx, Klds + (size_t)j * K2_DP + d);
    copy16_to_lds(Vg + idx, Vlds + (size_t)j * K2_DP + d);
  }
  async_copy_wait();
  __syncthreads();

  const int lane = tid & 31;
  const int waveId = tid >> 5;
  const int m = lane & 15;
  const int half = lane >> 4;
  _Float16* pw = Plds + waveId * (16 * K2_PP);
  const float SCALE = 0.125f;  // 64^-0.5

  for (int itile = waveId; itile < 16; itile += 8) {
    const _Float16* qrow = Qws + (size_t)w * NTOK * DH;
    v16h qa0 = load_a_frag_f16(qrow, DH, itile * 16 + m, half, 0);
    v16h qa1 = load_a_frag_f16(qrow, DH, itile * 16 + m, half, 32);

    v8f o0 = {}, o1 = {}, o2 = {}, o3 = {};
    float mrow[8], lrow[8];
#pragma unroll
    for (int r = 0; r < 8; ++r) { mrow[r] = -1e30f; lrow[r] = 0.0f; }

    for (int jp = 0; jp < 16; ++jp) {     // 32 kv tokens per step
      const int j0 = jp * 32;
      // Issue all 4 K-fragment LDS loads before the WMMAs.
      v16h kb00 = load_b_frag_f16(Klds, K2_DP, j0 + m,      half, 0);
      v16h kb01 = load_b_frag_f16(Klds, K2_DP, j0 + m,      half, 32);
      v16h kb10 = load_b_frag_f16(Klds, K2_DP, j0 + 16 + m, half, 0);
      v16h kb11 = load_b_frag_f16(Klds, K2_DP, j0 + 16 + m, half, 32);
      v8f s0 = {}, s1 = {};
      s0 = wmma16(qa0, kb00, s0);
      s0 = wmma16(qa1, kb01, s0);
      s1 = wmma16(qa0, kb10, s1);
      s1 = wmma16(qa1, kb11, s1);

      float alpha[8];
#pragma unroll
      for (int r = 0; r < 8; ++r) {
        float a  = s0[r] * SCALE;
        float b2 = s1[r] * SCALE;
        float mx = fmaxf(a, b2);
        mx = fmaxf(mx, __shfl_xor(mx, 1, 32));
        mx = fmaxf(mx, __shfl_xor(mx, 2, 32));
        mx = fmaxf(mx, __shfl_xor(mx, 4, 32));
        mx = fmaxf(mx, __shfl_xor(mx, 8, 32));
        float nm = fmaxf(mrow[r], mx);
        float al = __expf(mrow[r] - nm);
        float p0 = __expf(a - nm);
        float p1 = __expf(b2 - nm);
        float ps = p0 + p1;
        ps += __shfl_xor(ps, 1, 32);
        ps += __shfl_xor(ps, 2, 32);
        ps += __shfl_xor(ps, 4, 32);
        ps += __shfl_xor(ps, 8, 32);
        lrow[r] = lrow[r] * al + ps;
        mrow[r] = nm;
        alpha[r] = al;
        // Bounce P through per-wave LDS scratch: C-layout -> A-layout.
        pw[(size_t)(r + half * 8) * K2_PP + m]      = (_Float16)p0;
        pw[(size_t)(r + half * 8) * K2_PP + 16 + m] = (_Float16)p1;
      }
#pragma unroll
      for (int r = 0; r < 8; ++r) {
        o0[r] *= alpha[r]; o1[r] *= alpha[r];
        o2[r] *= alpha[r]; o3[r] *= alpha[r];
      }
      v16h pa = load_a_frag_f16(pw, K2_PP, m, half, 0);
      // V B-fragments gathered row-major from [j][d] (k = kv token).
      v16h vb0 = load_b_frag_f16_rm(Vlds + j0 * K2_DP, K2_DP,  0 + m, half, 0);
      v16h vb1 = load_b_frag_f16_rm(Vlds + j0 * K2_DP, K2_DP, 16 + m, half, 0);
      v16h vb2 = load_b_frag_f16_rm(Vlds + j0 * K2_DP, K2_DP, 32 + m, half, 0);
      v16h vb3 = load_b_frag_f16_rm(Vlds + j0 * K2_DP, K2_DP, 48 + m, half, 0);
      o0 = wmma16(pa, vb0, o0);
      o1 = wmma16(pa, vb1, o1);
      o2 = wmma16(pa, vb2, o2);
      o3 = wmma16(pa, vb3, o3);
    }

    _Float16* ao = AOws + (size_t)w * NTOK * DH;
#pragma unroll
    for (int r = 0; r < 8; ++r) {
      float inv = 1.0f / lrow[r];
      size_t row = (size_t)(itile * 16 + r + half * 8);
      ao[row * DH +  0 + m] = (_Float16)(o0[r] * inv);
      ao[row * DH + 16 + m] = (_Float16)(o1[r] * inv);
      ao[row * DH + 32 + m] = (_Float16)(o2[r] * inv);
      ao[row * DH + 48 + m] = (_Float16)(o3[r] * inv);
    }
  }
}

// ---------------------------------------------------------------------------
// Kernel 3: output projection out = Wo * AO + bo. Block = (batch, window).
// LDS: gathered AO for all 8 heads [256 tok][pitch 520] = 266KB dynamic,
// async-b128 staged.
// ---------------------------------------------------------------------------
#define K3_CP 520

__global__ void out_proj_kernel(const _Float16* __restrict__ AOws,
                                const float* __restrict__ Wo,
                                const float* __restrict__ bo,
                                float* __restrict__ out) {
  extern __shared__ _Float16 lds[];
  const int bx  = blockIdx.x;          // b*64 + win
  const int b   = bx >> 6;
  const int win = bx & 63;
  const int Xb  = win >> 3, Yb = win & 7;
  const int tid = threadIdx.x;

  // Gather 8 heads of this window: lds[tok][head*64 + d].
  for (int head = 0; head < 8; ++head) {
    size_t w = (size_t)(((b * 8 + head) * 8 + Xb) * 8 + Yb);
    const _Float16* src = AOws + w * NTOK * DH;
#pragma unroll
    for (int it = 0; it < 8; ++it) {
      int idx = (it * 256 + tid) * 8;
      int tok = idx >> 6, d = idx & 63;
      copy16_to_lds(src + idx, lds + (size_t)tok * K3_CP + head * 64 + d);
    }
  }
  async_copy_wait();
  __syncthreads();

  const int lane = tid & 31;
  const int waveId = tid >> 5;
  const int m = lane & 15;
  const int half = lane >> 4;

  for (int t = waveId; t < 256; t += 8) {   // 16 m-tiles x 16 n-tiles
    const int mt = t >> 4, nt = t & 15;
    const float* wrow = Wo + (size_t)(mt * 16) * 512;
    __builtin_prefetch(wrow + (size_t)m * 512, 0, 1);
    v8f acc = {};
#pragma unroll
    for (int kc = 0; kc < 16; ++kc) {
      v16h a  = load_a_frag_f32(wrow, 512, m, half, kc * 32);
      v16h bb = load_b_frag_f16(lds, K3_CP, nt * 16 + m, half, kc * 32);
      acc = wmma16(a, bb, acc);
    }
    const int tok = nt * 16 + m;
    const int hh  = Xb * WIN + (tok >> 4);
    const int ww  = Yb * WIN + (tok & 15);
    const size_t pix = (size_t)hh * WWID + ww;
#pragma unroll
    for (int r = 0; r < 8; ++r) {
      int o = mt * 16 + r + half * 8;
      out[((size_t)b * 256 + o) * (HH * WWID) + pix] = acc[r] + bo[o];
    }
  }
}

// ---------------------------------------------------------------------------
// Launch
// ---------------------------------------------------------------------------
extern "C" void kernel_launch(void* const* d_in, const int* in_sizes, int n_in,
                              void* d_out, int out_size, void* d_ws, size_t ws_size,
                              hipStream_t stream) {
  const float* x    = (const float*)d_in[0];
  const float* skip = (const float*)d_in[1];
  const float* Wq   = (const float*)d_in[2];
  const float* Wkv  = (const float*)d_in[3];
  const float* Wo   = (const float*)d_in[4];
  const float* bo   = (const float*)d_in[5];
  float* out = (float*)d_out;

  _Float16* Qws  = (_Float16*)d_ws;                      // 1024*256*64
  _Float16* Kws  = Qws + (size_t)NWIN * NTOK * DH;       // 1024*512*64
  _Float16* Vws  = Kws + (size_t)NWIN * NKV * DH;        // 1024*512*64
  _Float16* AOws = Vws + (size_t)NWIN * NKV * DH;        // 1024*256*64

  const size_t lds1 = (size_t)NTOK * K1_PITCH * sizeof(_Float16);
  const size_t lds2 = (size_t)(2 * NKV * K2_DP + 8 * 16 * K2_PP) *
                      sizeof(_Float16);
  const size_t lds3 = (size_t)NTOK * K3_CP * sizeof(_Float16);

  qkv_proj_kernel<<<256, 256, lds1, stream>>>(x, skip, Wq, Wkv, Qws, Kws, Vws);
  attn_kernel<<<NWIN, 256, lds2, stream>>>(Qws, Kws, Vws, AOws);
  out_proj_kernel<<<128, 256, lds3, stream>>>(AOws, Wo, bo, out);
}
